// DurationModel_46162308497828
// MI455X (gfx1250) — compile-verified, matrix operand order checked by reading, not run on hardware
//
#include <hip/hip_runtime.h>
#include <hip/hip_bf16.h>

// ---------------------------------------------------------------------------
// Bidirectional LSTM duration model for MI455X (gfx1250, wave32, WMMA).
//   B=32, L=2048, D=256, gates 4D=1024.
//   1) prep_weights : f32 w[512,1024] -> bf16 wx^T / wh^T [1024,256]
//   2) xg_gemm      : xg = embed[phonemes] @ wx + b, WMMA bf16, embedding
//                     gather fused into LDS A staging, B double-buffered
//                     (pinned with sched_barrier)
//   3) lstm         : sequential recurrence, WMMA bf16 for h @ wh; gate
//                     quadruple per-lane; c in registers; h via padded LDS;
//                     wh B double-buffered + sched_barrier; xg async-DMA'd
//                     ahead into double-buffered LDS (ASYNCcnt)
//   4) proj         : relu([h_fwd, flip(h_bwd)] @ w_proj + b_proj)
// ---------------------------------------------------------------------------

#define B_   32
#define L_   2048
#define D_   256
#define ND_  1024   // 4*D
#define HPAD 264    // 256 + 8 bf16 pad -> row stride 528B (16B aligned, bank-spread)

typedef __attribute__((ext_vector_type(16))) __bf16 v16bf;
typedef __attribute__((ext_vector_type(8)))  float  v8f;
typedef __attribute__((ext_vector_type(4)))  int    i32x4;

__device__ __forceinline__ unsigned short f2bf(float f) {
  union { float f; unsigned u; } v; v.f = f;
  unsigned r = v.u + 0x7FFFu + ((v.u >> 16) & 1u);   // round-to-nearest-even
  return (unsigned short)(r >> 16);
}
__device__ __forceinline__ float bf2f(unsigned short h) {
  union { unsigned u; float f; } v; v.u = ((unsigned)h) << 16;
  return v.f;
}
__device__ __forceinline__ v8f vsplat(float x) {
  v8f v;
#pragma unroll
  for (int i = 0; i < 8; ++i) v[i] = x;
  return v;
}

union TileBF { v16bf v; uint4 q[2]; };

// 16 bf16 operand slice for this lane: two b128 loads (K low/high halves).
__device__ __forceinline__ v16bf load_tile16(const unsigned short* p) {
  TileBF t;
  t.q[0] = *(const uint4*)(p);
  t.q[1] = *(const uint4*)(p + 16);
  return t.v;
}

__device__ __forceinline__ float sigmoidf_(float x) {
  return 1.0f / (1.0f + __expf(-x));
}
// tanh(x) = 1 - 2/(1+exp(2x)); saturates correctly at +/-inf, one transcendental
__device__ __forceinline__ float fast_tanhf_(float x) {
  return 1.0f - 2.0f / (1.0f + __expf(2.0f * x));
}

// Pin instruction groups: nothing crosses this point in the scheduler.
__device__ __forceinline__ void sched_fence() {
#if __has_builtin(__builtin_amdgcn_sched_barrier)
  __builtin_amdgcn_sched_barrier(0);
#endif
}

// ---- CDNA5 async global->LDS copy (ASYNCcnt-tracked), with safe fallback ---
__device__ __forceinline__ void async_copy_b128(const unsigned short* gsrc,
                                                unsigned short* ldst) {
#if __has_builtin(__builtin_amdgcn_global_load_async_to_lds_b128)
  __builtin_amdgcn_global_load_async_to_lds_b128(
      (__attribute__((address_space(1))) i32x4*)gsrc,
      (__attribute__((address_space(3))) i32x4*)ldst,
      0, 0);
#else
  *(uint4*)ldst = *(const uint4*)gsrc;   // sync fallback (VGPR bounce)
#endif
}

__device__ __forceinline__ void wait_async0() {
#if __has_builtin(__builtin_amdgcn_s_wait_asynccnt)
  __builtin_amdgcn_s_wait_asynccnt(0);
#else
  asm volatile("s_wait_asynccnt 0x0" ::: "memory");
#endif
}

// --------------------------------------------------------------------------
// 1) weight prep: w f32 [512,1024] -> wx^T, wh^T bf16 [1024,256]
// --------------------------------------------------------------------------
__global__ void __launch_bounds__(256)
prep_weights_kernel(const float* __restrict__ w,
                    unsigned short* __restrict__ wxt,
                    unsigned short* __restrict__ wht) {
  int idx = blockIdx.x * 256 + threadIdx.x;   // enumerates k*1024 + n, k<256
  int k = idx >> 10;
  int n = idx & 1023;
  wxt[(size_t)n * D_ + k] = f2bf(w[(size_t)k * ND_ + n]);
  wht[(size_t)n * D_ + k] = f2bf(w[(size_t)(D_ + k) * ND_ + n]);
}

// --------------------------------------------------------------------------
// 2) xg = embed[phonemes] @ wx + bias (bf16 out). Block = 16 waves, one
//    16-row M tile staged in LDS (embedding gather fused); each wave owns
//    4 N-tiles; B tiles double-buffered + sched-pinned. rev=1 -> reversed.
// --------------------------------------------------------------------------
__global__ void __launch_bounds__(512)
xg_gemm_kernel(const int* __restrict__ phonemes,
               const float* __restrict__ embed,
               const unsigned short* __restrict__ wt,    // [1024,256] bf16 (wx^T)
               const float* __restrict__ bias,           // [1024]
               unsigned short* __restrict__ xg,          // [B*L,1024] bf16
               int rev) {
  __shared__ __align__(16) unsigned short a_lds[16 * HPAD];

  const int tid  = threadIdx.x;
  const int wave = tid >> 5;               // 0..15
  const int lane = tid & 31;
  const int lh   = lane & 15;
  const int hi   = lane >> 4;

  const int i0 = blockIdx.x * 16;          // scan-row base (b*L + s0)
  const int b  = i0 >> 11;                 // / L_
  const int s0 = i0 & (L_ - 1);

  // Stage A tile: 512 chunks of 8 elements, one per thread.
  {
    int c    = tid;
    int m    = c >> 5;
    int k0   = (c & 31) * 8;
    int sA   = s0 + m;
    int srcs = rev ? (L_ - 1 - sA) : sA;
    int tok  = phonemes[b * L_ + srcs];
    const float* e = embed + (size_t)tok * D_ + k0;
    float4 f0 = *(const float4*)(e);
    float4 f1 = *(const float4*)(e + 4);
    union { uint4 q; unsigned short us[8]; } t;
    t.us[0] = f2bf(f0.x); t.us[1] = f2bf(f0.y);
    t.us[2] = f2bf(f0.z); t.us[3] = f2bf(f0.w);
    t.us[4] = f2bf(f1.x); t.us[5] = f2bf(f1.y);
    t.us[6] = f2bf(f1.z); t.us[7] = f2bf(f1.w);
    *(uint4*)&a_lds[m * HPAD + k0] = t.q;
  }
  __syncthreads();

  const int nt0 = wave * 4;
  const unsigned short* brow[4];
  v8f acc[4];
#pragma unroll
  for (int q = 0; q < 4; ++q) {
    int n = (nt0 + q) * 16 + lh;
    acc[q]  = vsplat(bias[n]);             // fuse bias into accumulator init
    brow[q] = wt + (size_t)n * D_;
  }

  // Double-buffered B tiles; sched_fence keeps kt+1 loads above kt WMMAs.
  v16bf bc[4], bnx[4];
#pragma unroll
  for (int q = 0; q < 4; ++q) bc[q] = load_tile16(brow[q] + hi * 8);

#pragma unroll
  for (int kt = 0; kt < 8; ++kt) {
    const int k0 = kt * 32 + hi * 8;
    if (kt < 7) {
      const int k1 = (kt + 1) * 32 + hi * 8;
#pragma unroll
      for (int q = 0; q < 4; ++q) bnx[q] = load_tile16(brow[q] + k1);
    }
    v16bf a = load_tile16(a_lds + lh * HPAD + k0);   // ds_load_b128 x2
    sched_fence();
#pragma unroll
    for (int q = 0; q < 4; ++q)
      acc[q] = __builtin_amdgcn_wmma_f32_16x16x32_bf16(
          false, a, false, bc[q], (short)0, acc[q], false, false);
    sched_fence();
#pragma unroll
    for (int q = 0; q < 4; ++q) bc[q] = bnx[q];      // register rename
  }

#pragma unroll
  for (int q = 0; q < 4; ++q) {
    int n = (nt0 + q) * 16 + lh;
#pragma unroll
    for (int r = 0; r < 8; ++r) {
      int m = r + hi * 8;                  // C layout: vgpr r -> M=r (+8 hi half)
      size_t row = (size_t)b * L_ + (size_t)(s0 + m);
      xg[row * ND_ + n] = f2bf(acc[q][r]);
    }
  }
}

// --------------------------------------------------------------------------
// 3) Sequential LSTM. Grid (bhalf=2, dir=2), block = 512 (16 waves).
//    Wave w owns gate channels j in [16w,16w+16): N-tiles at n = 256q+16w
//    for q = i,g,f,o -> each lane holds its full gate quadruple; c in
//    registers; h via padded LDS; wh B double-buffered + sched-pinned;
//    kt=0 peeled with inline-zero C; step s+1's xg tile async-DMA'd into
//    double-buffered LDS during step s.
// --------------------------------------------------------------------------
__global__ void __launch_bounds__(512)
lstm_kernel(const unsigned short* __restrict__ xg_f,
            const unsigned short* __restrict__ xg_b,
            const unsigned short* __restrict__ wh_f,   // [1024,256] bf16 (wh^T)
            const unsigned short* __restrict__ wh_b,
            const int* __restrict__ lengths,
            unsigned short* __restrict__ h_f,          // [B*L,256] bf16 scan order
            unsigned short* __restrict__ h_b) {
  __shared__ __align__(16) unsigned short h_lds[16 * HPAD];       //  8448 B
  __shared__ __align__(16) unsigned short xg_lds[2][16 * ND_];    // 65536 B

  const int bhalf = blockIdx.x;
  const int dir   = blockIdx.y;
  const int b0    = bhalf * 16;
  const unsigned short* xg = dir ? xg_b : xg_f;
  const unsigned short* wt = dir ? wh_b : wh_f;
  unsigned short* hout     = dir ? h_b : h_f;

  const int tid  = threadIdx.x;
  const int wave = tid >> 5;
  const int lane = tid & 31;
  const int lh   = lane & 15;
  const int hi   = lane >> 4;
  const int j    = wave * 16 + lh;         // owned hidden channel 0..255

  for (int i = tid; i < 16 * HPAD; i += 512) h_lds[i] = 0;

  const unsigned short* wrow[4];
#pragma unroll
  for (int q = 0; q < 4; ++q) wrow[q] = wt + (size_t)(q * D_ + j) * D_;

  float c[8];
  int   len8[8];
#pragma unroll
  for (int r = 0; r < 8; ++r) {
    c[r]    = 0.0f;
    len8[r] = lengths[b0 + r + hi * 8];
  }

  const v8f czero = vsplat(0.0f);          // folds to WMMA inline SRC2=0

  // Prologue: DMA xg tile for step 0 into buffer 0 (wave w loads row w).
  {
    const size_t grow = ((size_t)(b0 + wave) * L_ + 0) * ND_;
    unsigned short* dst = &xg_lds[0][wave * ND_];
#pragma unroll
    for (int i = 0; i < 4; ++i)
      async_copy_b128(xg + grow + i * 256 + lane * 8, dst + i * 256 + lane * 8);
  }
  wait_async0();
  __syncthreads();

  for (int s = 0; s < L_; ++s) {
    v8f acc[4];
    v16bf bc[4], bnx[4];

    // kt = 0 peeled: C = inline zero; preload kt=1 B tiles before WMMAs.
#pragma unroll
    for (int q = 0; q < 4; ++q) bc[q] = load_tile16(wrow[q] + hi * 8);
    {
      const int k1 = 32 + hi * 8;
#pragma unroll
      for (int q = 0; q < 4; ++q) bnx[q] = load_tile16(wrow[q] + k1);
      v16bf a = load_tile16(h_lds + lh * HPAD + hi * 8);
      sched_fence();
#pragma unroll
      for (int q = 0; q < 4; ++q)
        acc[q] = __builtin_amdgcn_wmma_f32_16x16x32_bf16(
            false, a, false, bc[q], (short)0, czero, false, false);
      sched_fence();
#pragma unroll
      for (int q = 0; q < 4; ++q) bc[q] = bnx[q];
    }

#pragma unroll
    for (int kt = 1; kt < 8; ++kt) {
      const int k0 = kt * 32 + hi * 8;
      if (kt < 7) {
        const int k1 = (kt + 1) * 32 + hi * 8;
#pragma unroll
        for (int q = 0; q < 4; ++q) bnx[q] = load_tile16(wrow[q] + k1);
      }
      v16bf a = load_tile16(h_lds + lh * HPAD + k0);   // A: carried h (LDS)
      sched_fence();
#pragma unroll
      for (int q = 0; q < 4; ++q)
        acc[q] = __builtin_amdgcn_wmma_f32_16x16x32_bf16(
            false, a, false, bc[q], (short)0, acc[q], false, false);
      sched_fence();
#pragma unroll
      for (int q = 0; q < 4; ++q) bc[q] = bnx[q];
    }

    // Kick off async DMA of step s+1's xg tile (other parity buffer).
    if (s + 1 < L_) {
      const size_t grow = ((size_t)(b0 + wave) * L_ + (s + 1)) * ND_;
      unsigned short* dst = &xg_lds[(s + 1) & 1][wave * ND_];
#pragma unroll
      for (int i = 0; i < 4; ++i)
        async_copy_b128(xg + grow + i * 256 + lane * 8, dst + i * 256 + lane * 8);
    }
    __syncthreads();   // everyone finished reading h_lds for this step

    const unsigned short* xb = &xg_lds[s & 1][0];
#pragma unroll
    for (int r = 0; r < 8; ++r) {
      const int m   = r + hi * 8;          // batch row within half
      const int bg_ = b0 + m;              // global batch row
      float gi = acc[0][r] + bf2f(xb[m * ND_ + 0 * D_ + j]);
      float gg = acc[1][r] + bf2f(xb[m * ND_ + 1 * D_ + j]);
      float gf = acc[2][r] + bf2f(xb[m * ND_ + 2 * D_ + j]);
      float go = acc[3][r] + bf2f(xb[m * ND_ + 3 * D_ + j]);

      float is = sigmoidf_(gi);
      float gt = fast_tanhf_(gg);
      float fs = sigmoidf_(gf + 1.0f);     // hk.LSTM forget bias
      float os = sigmoidf_(go);
      float cn = fs * c[r] + is * gt;
      float hv = os * fast_tanhf_(cn);

      hout[((size_t)bg_ * L_ + s) * D_ + j] = f2bf(hv);

      // ResetCore: zero carry entering step s+1 where mask_bwd[:, s+1] true
      int rn = dir ? ((L_ - 2 - s) < (len8[r] - 1)) : 0;
      c[r] = rn ? 0.0f : cn;
      h_lds[m * HPAD + j] = rn ? (unsigned short)0 : f2bf(hv);
    }

    wait_async0();     // this wave's DMA for s+1 done (visible after barrier)
    __syncthreads();   // h_lds + xg_lds[(s+1)&1] ready for next step
  }
}

// --------------------------------------------------------------------------
// 4) projection: out[b,t] = relu(h_fwd[b,t]·wp[:256] + h_bwd[b,L-1-t]·wp[256:] + bp)
//    One wave per output element, shuffle reduction.
// --------------------------------------------------------------------------
__global__ void __launch_bounds__(128)
proj_kernel(const unsigned short* __restrict__ hf,
            const unsigned short* __restrict__ hb,
            const float* __restrict__ wp,     // [512]
            const float* __restrict__ bp,     // [1]
            float* __restrict__ out) {        // [B*L]
  const int gid  = blockIdx.x * 4 + (threadIdx.x >> 5);
  const int lane = threadIdx.x & 31;
  const int b = gid >> 11;
  const int t = gid & (L_ - 1);
  const size_t rf = ((size_t)b * L_ + t) * D_;
  const size_t rb = ((size_t)b * L_ + (L_ - 1 - t)) * D_;

  float sum = 0.0f;
#pragma unroll
  for (int k = 0; k < 8; ++k) {
    int jj = lane + 32 * k;
    sum += bf2f(hf[rf + jj]) * wp[jj];
    sum += bf2f(hb[rb + jj]) * wp[D_ + jj];
  }
#pragma unroll
  for (int off = 16; off > 0; off >>= 1) sum += __shfl_down(sum, off, 32);
  if (lane == 0) {
    float y = sum + bp[0];
    out[gid] = y > 0.0f ? y : 0.0f;
  }
}

// --------------------------------------------------------------------------
extern "C" void kernel_launch(void* const* d_in, const int* in_sizes, int n_in,
                              void* d_out, int out_size, void* d_ws, size_t ws_size,
                              hipStream_t stream) {
  const int*   phonemes = (const int*)d_in[0];
  const int*   lengths  = (const int*)d_in[1];
  const float* embed    = (const float*)d_in[2];
  const float* w_fwd    = (const float*)d_in[3];
  const float* b_fwd    = (const float*)d_in[4];
  const float* w_bwd    = (const float*)d_in[5];
  const float* b_bwd    = (const float*)d_in[6];
  const float* w_proj   = (const float*)d_in[7];
  const float* b_proj   = (const float*)d_in[8];
  float* out = (float*)d_out;

  char* ws = (char*)d_ws;
  size_t off = 0;
  auto wsalloc = [&](size_t bytes) -> char* {
    char* p = ws + off;
    off += (bytes + 255) & ~(size_t)255;
    return p;
  };
  unsigned short* wxt_f = (unsigned short*)wsalloc((size_t)ND_ * D_ * 2);
  unsigned short* wht_f = (unsigned short*)wsalloc((size_t)ND_ * D_ * 2);
  unsigned short* wxt_b = (unsigned short*)wsalloc((size_t)ND_ * D_ * 2);
  unsigned short* wht_b = (unsigned short*)wsalloc((size_t)ND_ * D_ * 2);
  unsigned short* xg_f  = (unsigned short*)wsalloc((size_t)B_ * L_ * ND_ * 2);
  unsigned short* xg_b  = (unsigned short*)wsalloc((size_t)B_ * L_ * ND_ * 2);
  unsigned short* h_f   = (unsigned short*)wsalloc((size_t)B_ * L_ * D_ * 2);
  unsigned short* h_b   = (unsigned short*)wsalloc((size_t)B_ * L_ * D_ * 2);

  prep_weights_kernel<<<(D_ * ND_) / 256, 256, 0, stream>>>(w_fwd, wxt_f, wht_f);
  prep_weights_kernel<<<(D_ * ND_) / 256, 256, 0, stream>>>(w_bwd, wxt_b, wht_b);
  xg_gemm_kernel<<<(B_ * L_) / 16, 512, 0, stream>>>(phonemes, embed, wxt_f, b_fwd, xg_f, 0);
  xg_gemm_kernel<<<(B_ * L_) / 16, 512, 0, stream>>>(phonemes, embed, wxt_b, b_bwd, xg_b, 1);
  lstm_kernel<<<dim3(2, 2), 512, 0, stream>>>(xg_f, xg_b, wht_f, wht_b, lengths, h_f, h_b);
  proj_kernel<<<(B_ * L_) / 4, 128, 0, stream>>>(h_f, h_b, w_proj, b_proj, out);
}
